// SmoothnessLoss_86766929314299
// MI455X (gfx1250) — matrix operand order repaired and checked by reading, not compiled, not run on hardware
//
#include <hip/hip_runtime.h>
#include <hip/hip_bf16.h>

typedef __attribute__((ext_vector_type(16))) __bf16       v16bf;
typedef __attribute__((ext_vector_type(8)))  float        v8f;
typedef __attribute__((ext_vector_type(4)))  unsigned int u32x4;

#define NB   8
#define NN   2048
#define DD   128
#define TSTR 68   // LDS row stride in dwords (64 data + 4 pad), 272B -> 16B aligned rows

static __device__ __forceinline__ unsigned int f2bf(float f) {
    unsigned int u = __float_as_uint(f);
    return (u + 0x7FFFu + ((u >> 16) & 1u)) >> 16;   // round-to-nearest-even bf16
}
static __device__ __forceinline__ unsigned int pack2(float lo, float hi) {
    return f2bf(lo) | (f2bf(hi) << 16);
}

union FragCast { u32x4 q[2]; v16bf v; };

__global__ __launch_bounds__(256)
void smoothness_main(const float* __restrict__ adj,
                     const float* __restrict__ emb,
                     float* __restrict__ ws) {
    extern __shared__ char smem_raw[];
    unsigned int* tA   = (unsigned int*)smem_raw;          // 128 x TSTR dwords
    unsigned int* tB   = tA + 128 * TSTR;                  // 128 x TSTR dwords
    float*        hsqA = (float*)(tB + 128 * TSTR);        // 128
    float*        hsqB = hsqA + 128;                       // 128
    float*        red  = hsqB + 128;                       // 8

    const int t  = threadIdx.x;
    const int b  = blockIdx.z;
    const int Ib = blockIdx.y * 128;   // m-strip base
    const int Jb = blockIdx.x * 128;   // n-strip base

    // ---- Stage node_emb (f32 -> bf16) into LDS, coalesced float4 loads ----
    const float4* eA = (const float4*)(emb + ((size_t)b * NN + Ib) * DD);
    const float4* eB = (const float4*)(emb + ((size_t)b * NN + Jb) * DD);
    #pragma unroll
    for (int i = 0; i < 16; ++i) {
        int idx = t + 256 * i;            // float4 index within 128x128 panel
        int row = idx >> 5;               // 32 float4 per row
        int c4  = idx & 31;
        int d0  = row * TSTR + c4 * 2;
        float4 va = eA[idx];
        tA[d0]     = pack2(va.x, va.y);
        tA[d0 + 1] = pack2(va.z, va.w);
        float4 vb = eB[idx];
        tB[d0]     = pack2(vb.x, vb.y);
        tB[d0 + 1] = pack2(vb.z, vb.w);
    }
    __syncthreads();

    // ---- Per-row squared norms (f32, from staged bf16), deterministic ----
    {
        int row = t & 127;
        const unsigned int* rp = ((t < 128) ? tA : tB) + row * TSTR;
        float s = 0.f;
        #pragma unroll
        for (int i = 0; i < 16; ++i) {
            u32x4 q = *(const u32x4*)(rp + i * 4);
            #pragma unroll
            for (int j = 0; j < 4; ++j) {
                unsigned int u = q[j];
                float f0 = __uint_as_float(u << 16);
                float f1 = __uint_as_float(u & 0xFFFF0000u);
                s = fmaf(f0, f0, s);
                s = fmaf(f1, f1, s);
            }
        }
        ((t < 128) ? hsqA : hsqB)[row] = s;
    }
    __syncthreads();

    // ---- WMMA gram + fused epilogue ----
    const int w = t >> 5;        // wave 0..7 -> 16-row strip of macro-tile
    const int l = t & 31;
    const int h = l >> 4;        // lane half
    const int m = l & 15;        // row (A) / col (B) within 16

    // Hoist the wave's A fragments for all K (ISA 16-bit A layout:
    // v0..3: K = kb + h*8 + 0..7 ; v4..7: K = kb + 16 + h*8 + 0..7)
    v16bf afr[4];
    {
        const unsigned int* arow = tA + (w * 16 + m) * TSTR;
        #pragma unroll
        for (int ks = 0; ks < 4; ++ks) {
            int kd = ks * 16;    // kb/2 dwords
            FragCast fc;
            fc.q[0] = *(const u32x4*)(arow + kd + h * 4);
            fc.q[1] = *(const u32x4*)(arow + kd + 8 + h * 4);
            afr[ks] = fc.v;
        }
    }

    // Per-lane row norms for the 8 C rows this lane touches (M = h*8 + r)
    float hm[8];
    {
        const float* hap = hsqA + w * 16 + h * 8;
        #pragma unroll
        for (int r = 0; r < 8; ++r) hm[r] = hap[r];
    }

    // adj base: row mG0 = Ib + w*16 + h*8, col nG = Jb + (l&15)
    const float* adjbase = adj + (((size_t)b * NN) + (Ib + w * 16 + h * 8)) * NN + Jb + m;

    float partial = 0.f;
    #pragma unroll 1
    for (int jt = 0; jt < 8; ++jt) {
        v8f acc = {0.f, 0.f, 0.f, 0.f, 0.f, 0.f, 0.f, 0.f};
        const unsigned int* brow = tB + (jt * 16 + m) * TSTR;
        #pragma unroll
        for (int ks = 0; ks < 4; ++ks) {
            // ISA 16-bit B layout: v0..3: K = kb + h*8 + 0..7 ; v4..7: K = kb + 16 + h*8 + 0..7
            int kd = ks * 16;
            FragCast fc;
            fc.q[0] = *(const u32x4*)(brow + kd + h * 4);
            fc.q[1] = *(const u32x4*)(brow + kd + 8 + h * 4);
            acc = __builtin_amdgcn_wmma_f32_16x16x32_bf16(
                    false, afr[ks], false, fc.v, (short)0, acc, false, false);
        }
        float hn = hsqB[jt * 16 + m];
        const float* ap = adjbase + jt * 16;
        #pragma unroll
        for (int r = 0; r < 8; ++r) {
            float d = fmaf(-2.f, acc[r], hm[r] + hn);   // ||hi||^2+||hj||^2-2g
            d = fmaxf(d, 0.f);
            partial = fmaf(ap[r * NN], d, partial);     // adj * sq_dist
        }
    }

    // ---- Deterministic reduction: lanes -> waves -> block partial ----
    #pragma unroll
    for (int off = 16; off > 0; off >>= 1)
        partial += __shfl_xor(partial, off, 32);
    if (l == 0) red[w] = partial;
    __syncthreads();
    if (t == 0) {
        float s = 0.f;
        #pragma unroll
        for (int i = 0; i < 8; ++i) s += red[i];
        int flat = blockIdx.x + 16 * (blockIdx.y + 16 * blockIdx.z);
        ws[flat] = s;
    }
}

__global__ __launch_bounds__(256)
void smoothness_reduce(const float* __restrict__ ws, float* __restrict__ out) {
    __shared__ float red[8];
    int t = threadIdx.x;
    float s = 0.f;
    for (int i = t; i < 2048; i += 256) s += ws[i];
    #pragma unroll
    for (int off = 16; off > 0; off >>= 1)
        s += __shfl_xor(s, off, 32);
    if ((t & 31) == 0) red[t >> 5] = s;
    __syncthreads();
    if (t == 0) {
        float tot = 0.f;
        #pragma unroll
        for (int i = 0; i < 8; ++i) tot += red[i];
        out[0] = tot / (float)(NB * NN);
    }
}

extern "C" void kernel_launch(void* const* d_in, const int* in_sizes, int n_in,
                              void* d_out, int out_size, void* d_ws, size_t ws_size,
                              hipStream_t stream) {
    const float* adj = (const float*)d_in[0];
    const float* emb = (const float*)d_in[1];
    float* ws  = (float*)d_ws;
    float* out = (float*)d_out;

    dim3 grid(16, 16, NB);     // (J, I, batch) -> 2048 blocks
    dim3 block(256);
    size_t shmem = (size_t)(2 * 128 * TSTR + 2 * 128 + 8) * sizeof(float); // ~70.7 KB

    smoothness_main<<<grid, block, shmem, stream>>>(adj, emb, ws);
    smoothness_reduce<<<1, 256, 0, stream>>>(ws, out);
}
